// ByteMoE_49314814492694
// MI455X (gfx1250) — compile-verified, matrix operand order checked by reading, not run on hardware
//
#include <hip/hip_runtime.h>
#include <math.h>

typedef __bf16 bf16_t;
typedef __attribute__((ext_vector_type(16))) __bf16 v16bf;
typedef __attribute__((ext_vector_type(8)))  float  v8f;
typedef __attribute__((ext_vector_type(4)))  int    v4i_;

// ---- problem sizes (fixed by the reference) ----
static constexpr int S    = 8192;   // B*T
static constexpr int D    = 1024;
static constexpr int F    = 4096;
static constexpr int E    = 8;
static constexpr int TOPK = 2;
static constexpr int CAP  = 2560;   // int(1.25 * S * K / E)
static constexpr int SK   = S * TOPK;

// ---- workspace layout (bytes) ----
static constexpr size_t OFF_IMP  = 0;                                   // E f32
static constexpr size_t OFF_CNT  = 256;                                 // E f32
static constexpr size_t OFF_IDX  = 512;                                 // SK i32
static constexpr size_t OFF_VAL  = OFF_IDX + (size_t)SK * 4;            // SK f32
static constexpr size_t OFF_POS  = OFF_VAL + (size_t)SK * 4;            // SK i32
static constexpr size_t OFF_W1B  = (OFF_POS + (size_t)SK * 4 + 255) & ~(size_t)255;
static constexpr size_t OFF_W2B  = OFF_W1B + (size_t)E * F * D * 2;     // bf16
static constexpr size_t OFF_BUFB = OFF_W2B + (size_t)E * D * F * 2;     // bf16 [E,CAP,D]
static constexpr size_t OFF_H    = OFF_BUFB + (size_t)E * CAP * D * 2;  // bf16 [E,CAP,F]
static constexpr size_t OFF_OUT2 = OFF_H + (size_t)E * CAP * F * 2;     // f32  [E,CAP,D]

struct bf16x4 { bf16_t a, b, c, d; };
union FragU { uint4 u[2]; v16bf v; };

// ---- gfx1250 async global->LDS copy (bypasses VGPRs, tracked by ASYNCcnt) ----
#if defined(__AMDGCN__) && __has_builtin(__builtin_amdgcn_global_load_async_to_lds_b128)
#define HAVE_ASYNC_CP 1
typedef __attribute__((address_space(1))) v4i_ as1_v4i;
typedef __attribute__((address_space(3))) v4i_ as3_v4i;
#else
#define HAVE_ASYNC_CP 0
#endif

__device__ __forceinline__ void cp16_async(const void* g, void* l) {
#if HAVE_ASYNC_CP
  __builtin_amdgcn_global_load_async_to_lds_b128(
      (as1_v4i*)(g), (as3_v4i*)(l), 0, 0);
#else
  (void)g; (void)l;
#endif
}

__device__ __forceinline__ void wait_async0() {
#if HAVE_ASYNC_CP
#if __has_builtin(__builtin_amdgcn_s_wait_asynccnt)
  __builtin_amdgcn_s_wait_asynccnt(0);
#else
  asm volatile("s_wait_asynccnt 0" ::: "memory");
#endif
#endif
}

// ---------------- small helpers ----------------
__global__ void zero_f32_kernel(float* p, int n) {
  int i = blockIdx.x * blockDim.x + threadIdx.x;
  if (i < n) p[i] = 0.0f;
}

__global__ void zero_b128_kernel(uint4* p, size_t n) {
  size_t i = blockIdx.x * (size_t)blockDim.x + threadIdx.x;
  if (i < n) { uint4 z; z.x = 0u; z.y = 0u; z.z = 0u; z.w = 0u; p[i] = z; }
}

__global__ void f32_to_bf16_kernel(const float* __restrict__ src,
                                   bf16_t* __restrict__ dst, size_t n4) {
  size_t i = blockIdx.x * (size_t)blockDim.x + threadIdx.x;
  if (i >= n4) return;
  float4 v = reinterpret_cast<const float4*>(src)[i];
  bf16x4 o; o.a = (bf16_t)v.x; o.b = (bf16_t)v.y; o.c = (bf16_t)v.z; o.d = (bf16_t)v.w;
  reinterpret_cast<bf16x4*>(dst)[i] = o;
}

// ---------------- router: logits + softmax + top2 + importance ----------------
__global__ __launch_bounds__(256)
void moe_router(const float* __restrict__ x, const float* __restrict__ wg,
                int* __restrict__ topk_idx, float* __restrict__ topk_val,
                float* __restrict__ imp) {
  __shared__ float wsm[E * D]; // 32 KB: full gate matrix
  int tid = threadIdx.x;
  for (int i = tid * 4; i < E * D; i += 256 * 4)
    *reinterpret_cast<float4*>(&wsm[i]) = *reinterpret_cast<const float4*>(&wg[i]);
  __syncthreads();

  int wave = tid >> 5, lane = tid & 31;
  int s = blockIdx.x * 8 + wave;
  const float* xr = x + (size_t)s * D;

  float acc[E];
#pragma unroll
  for (int e = 0; e < E; e++) acc[e] = 0.0f;

  for (int it = 0; it < D / 128; ++it) {
    int d = it * 128 + lane * 4;
    float4 xv = *reinterpret_cast<const float4*>(&xr[d]);
#pragma unroll
    for (int e = 0; e < E; e++) {
      float4 wv = *reinterpret_cast<const float4*>(&wsm[e * D + d]);
      acc[e] += xv.x * wv.x + xv.y * wv.y + xv.z * wv.z + xv.w * wv.w;
    }
  }
#pragma unroll
  for (int e = 0; e < E; e++)
    for (int off = 16; off >= 1; off >>= 1)
      acc[e] += __shfl_xor(acc[e], off, 32);

  if (lane == 0) {
    float mx = acc[0];
#pragma unroll
    for (int e = 1; e < E; e++) mx = fmaxf(mx, acc[e]);
    float p[E]; float sum = 0.0f;
#pragma unroll
    for (int e = 0; e < E; e++) { p[e] = expf(acc[e] - mx); sum += p[e]; }
    float inv = 1.0f / sum;
#pragma unroll
    for (int e = 0; e < E; e++) { p[e] *= inv; atomicAdd(&imp[e], p[e]); }
    // top-2 with first-occurrence tie break (matches lax.top_k)
    int i0 = 0;
    for (int e = 1; e < E; e++) if (p[e] > p[i0]) i0 = e;
    int i1 = (i0 == 0) ? 1 : 0;
    for (int e = 0; e < E; e++) if (e != i0 && p[e] > p[i1]) i1 = e;
    topk_idx[s * 2 + 0] = i0; topk_val[s * 2 + 0] = p[i0];
    topk_idx[s * 2 + 1] = i1; topk_val[s * 2 + 1] = p[i1];
  }
}

// ---------------- deterministic arrival-order scan ----------------
__global__ __launch_bounds__(256)
void moe_scan(const int* __restrict__ topk_idx, int* __restrict__ pos,
              float* __restrict__ counts) {
  __shared__ int lcnt[256][E];
  int tid = threadIdx.x;
  const int PER = SK / 256; // 64
  int base = tid * PER;
  int cnt[E];
#pragma unroll
  for (int e = 0; e < E; e++) cnt[e] = 0;
  for (int i = 0; i < PER; i++) cnt[topk_idx[base + i]]++;
#pragma unroll
  for (int e = 0; e < E; e++) lcnt[tid][e] = cnt[e];
  __syncthreads();
  if (tid < E) {
    int run = 0;
    for (int t = 0; t < 256; t++) { int c = lcnt[t][tid]; lcnt[t][tid] = run; run += c; }
    counts[tid] = (float)run;
  }
  __syncthreads();
  int off[E];
#pragma unroll
  for (int e = 0; e < E; e++) off[e] = lcnt[tid][e];
  for (int i = 0; i < PER; i++) { int e = topk_idx[base + i]; pos[base + i] = off[e]++; }
}

// ---------------- dispatch tokens -> bf16 [E,CAP,D] ----------------
__global__ __launch_bounds__(256)
void moe_dispatch(const float* __restrict__ x, const int* __restrict__ topk_idx,
                  const int* __restrict__ pos, bf16_t* __restrict__ bufb) {
  int a = blockIdx.x;
  int p = pos[a];
  if (p >= CAP) return;
  int e = topk_idx[a];
  int t = a >> 1;
  const float* src = x + (size_t)t * D;
  bf16_t* dst = bufb + ((size_t)e * CAP + p) * D;
  int d = threadIdx.x * 4;
  float4 v = *reinterpret_cast<const float4*>(&src[d]);
  bf16x4 o; o.a = (bf16_t)v.x; o.b = (bf16_t)v.y; o.c = (bf16_t)v.z; o.d = (bf16_t)v.w;
  *reinterpret_cast<bf16x4*>(&dst[d]) = o;
}

// ---------------- GEMM1: H = gelu(buf @ W1^T + b1), bf16 out ----------------
__global__ __launch_bounds__(256)
void moe_gemm1(const bf16_t* __restrict__ Abase, const bf16_t* __restrict__ Bbase,
               const float* __restrict__ b1, bf16_t* __restrict__ H) {
  constexpr int BM = 128, BK = 32, LDT = 40; // padded LDS stride (80B) -> conflict-free b128
  constexpr int TSZ = BM * LDT;              // one tile buffer (bf16 units)
  __shared__ __align__(16) bf16_t smem[4 * TSZ]; // 40 KB: A0,A1,B0,B1
  int e = blockIdx.z;
  int m0 = blockIdx.x * BM;
  int n0 = blockIdx.y * BM;
  const bf16_t* A = Abase + (size_t)e * CAP * D; // M=CAP, K=D (row-major, K contiguous)
  const bf16_t* B = Bbase + (size_t)e * F * D;   // N=F,   K=D (row-major, K contiguous)
  int tid = threadIdx.x, lane = tid & 31, wave = tid >> 5;
  int wm = wave & 1, wn = wave >> 1;

  v8f zero; for (int q = 0; q < 8; q++) zero[q] = 0.0f;
  v8f acc[4][2];
  for (int i = 0; i < 4; i++) for (int j = 0; j < 2; j++) acc[i][j] = zero;

  auto issueTile = [&](int kt, int bufId) {
    int k0 = kt * BK;
    bf16_t* As = smem + bufId * TSZ;
    bf16_t* Bs = smem + (2 + bufId) * TSZ;
#pragma unroll
    for (int c = 0; c < 2; c++) {
      int chunk = tid + c * 256;      // 0..511
      int row = chunk >> 2;
      int kc = (chunk & 3) * 8;       // bf16 units, 16B granules
#if HAVE_ASYNC_CP
      cp16_async(&A[((size_t)(m0 + row)) * D + k0 + kc], &As[row * LDT + kc]);
      cp16_async(&B[((size_t)(n0 + row)) * D + k0 + kc], &Bs[row * LDT + kc]);
#else
      uint4 av = *reinterpret_cast<const uint4*>(&A[((size_t)(m0 + row)) * D + k0 + kc]);
      *reinterpret_cast<uint4*>(&As[row * LDT + kc]) = av;
      uint4 bv = *reinterpret_cast<const uint4*>(&B[((size_t)(n0 + row)) * D + k0 + kc]);
      *reinterpret_cast<uint4*>(&Bs[row * LDT + kc]) = bv;
#endif
    }
  };

  issueTile(0, 0);
  const int NK = D / BK; // 32
  int half = lane >> 4, lrow = lane & 15;
  for (int kt = 0; kt < NK; ++kt) {
    wait_async0();
    __syncthreads();
    if (kt + 1 < NK) issueTile(kt + 1, (kt + 1) & 1);
    const bf16_t* As = smem + (kt & 1) * TSZ;
    const bf16_t* Bs = smem + (2 + (kt & 1)) * TSZ;
    FragU afr[4], bfr[2];
#pragma unroll
    for (int i = 0; i < 4; i++) {
      int r = wm * 64 + i * 16 + lrow;
      afr[i].u[0] = *reinterpret_cast<const uint4*>(&As[r * LDT + half * 8]);
      afr[i].u[1] = *reinterpret_cast<const uint4*>(&As[r * LDT + 16 + half * 8]);
    }
#pragma unroll
    for (int j = 0; j < 2; j++) {
      int r = wn * 32 + j * 16 + lrow;
      bfr[j].u[0] = *reinterpret_cast<const uint4*>(&Bs[r * LDT + half * 8]);
      bfr[j].u[1] = *reinterpret_cast<const uint4*>(&Bs[r * LDT + 16 + half * 8]);
    }
#pragma unroll
    for (int i = 0; i < 4; i++)
#pragma unroll
      for (int j = 0; j < 2; j++)
        acc[i][j] = __builtin_amdgcn_wmma_f32_16x16x32_bf16(
            false, afr[i].v, false, bfr[j].v, (short)0, acc[i][j], false, false);
  }

  // epilogue: +b1, exact gelu -> stage bf16 tile in LDS -> coalesced b128 stores
  constexpr int LDO = 136; // bf16 units; 272B row stride, 16B aligned
  bf16_t* ot = smem;       // 128*136*2 = 34816 B <= 40960 B
  __syncthreads();         // everyone done reading the last K tile
#pragma unroll
  for (int i = 0; i < 4; i++) {
#pragma unroll
    for (int j = 0; j < 2; j++) {
      int col = wn * 32 + j * 16 + lrow;
      float bb = b1[(size_t)e * F + n0 + col];
#pragma unroll
      for (int r = 0; r < 8; r++) {
        int row = wm * 64 + i * 16 + r + half * 8;
        float v = acc[i][j][r] + bb;
        float g = 0.5f * v * (1.0f + erff(v * 0.70710678118654752f));
        ot[row * LDO + col] = (bf16_t)g;
      }
    }
  }
  __syncthreads();
  for (int c = tid; c < BM * BM / 8; c += 256) { // 16B chunks
    int row = c >> 4;
    int col = (c & 15) * 8;
    uint4 v = *reinterpret_cast<const uint4*>(&ot[row * LDO + col]);
    *reinterpret_cast<uint4*>(&H[((size_t)e * CAP + m0 + row) * F + n0 + col]) = v;
  }
}

// ---------------- GEMM2: out2 = H @ W2^T + b2, f32 out ----------------
__global__ __launch_bounds__(256)
void moe_gemm2(const bf16_t* __restrict__ Abase, const bf16_t* __restrict__ Bbase,
               const float* __restrict__ b2, float* __restrict__ out2) {
  constexpr int BM = 128, BK = 32, LDT = 40;
  constexpr int TSZ = BM * LDT;
  __shared__ __align__(16) bf16_t smem[4 * TSZ];
  int e = blockIdx.z;
  int m0 = blockIdx.x * BM;
  int n0 = blockIdx.y * BM;
  const bf16_t* A = Abase + (size_t)e * CAP * F; // M=CAP, K=F
  const bf16_t* B = Bbase + (size_t)e * D * F;   // N=D,   K=F
  int tid = threadIdx.x, lane = tid & 31, wave = tid >> 5;
  int wm = wave & 1, wn = wave >> 1;

  v8f zero; for (int q = 0; q < 8; q++) zero[q] = 0.0f;
  v8f acc[4][2];
  for (int i = 0; i < 4; i++) for (int j = 0; j < 2; j++) acc[i][j] = zero;

  auto issueTile = [&](int kt, int bufId) {
    int k0 = kt * BK;
    bf16_t* As = smem + bufId * TSZ;
    bf16_t* Bs = smem + (2 + bufId) * TSZ;
#pragma unroll
    for (int c = 0; c < 2; c++) {
      int chunk = tid + c * 256;
      int row = chunk >> 2;
      int kc = (chunk & 3) * 8;
#if HAVE_ASYNC_CP
      cp16_async(&A[((size_t)(m0 + row)) * F + k0 + kc], &As[row * LDT + kc]);
      cp16_async(&B[((size_t)(n0 + row)) * F + k0 + kc], &Bs[row * LDT + kc]);
#else
      uint4 av = *reinterpret_cast<const uint4*>(&A[((size_t)(m0 + row)) * F + k0 + kc]);
      *reinterpret_cast<uint4*>(&As[row * LDT + kc]) = av;
      uint4 bv = *reinterpret_cast<const uint4*>(&B[((size_t)(n0 + row)) * F + k0 + kc]);
      *reinterpret_cast<uint4*>(&Bs[row * LDT + kc]) = bv;
#endif
    }
  };

  issueTile(0, 0);
  const int NK = F / BK; // 128
  int half = lane >> 4, lrow = lane & 15;
  for (int kt = 0; kt < NK; ++kt) {
    wait_async0();
    __syncthreads();
    if (kt + 1 < NK) issueTile(kt + 1, (kt + 1) & 1);
    const bf16_t* As = smem + (kt & 1) * TSZ;
    const bf16_t* Bs = smem + (2 + (kt & 1)) * TSZ;
    FragU afr[4], bfr[2];
#pragma unroll
    for (int i = 0; i < 4; i++) {
      int r = wm * 64 + i * 16 + lrow;
      afr[i].u[0] = *reinterpret_cast<const uint4*>(&As[r * LDT + half * 8]);
      afr[i].u[1] = *reinterpret_cast<const uint4*>(&As[r * LDT + 16 + half * 8]);
    }
#pragma unroll
    for (int j = 0; j < 2; j++) {
      int r = wn * 32 + j * 16 + lrow;
      bfr[j].u[0] = *reinterpret_cast<const uint4*>(&Bs[r * LDT + half * 8]);
      bfr[j].u[1] = *reinterpret_cast<const uint4*>(&Bs[r * LDT + 16 + half * 8]);
    }
#pragma unroll
    for (int i = 0; i < 4; i++)
#pragma unroll
      for (int j = 0; j < 2; j++)
        acc[i][j] = __builtin_amdgcn_wmma_f32_16x16x32_bf16(
            false, afr[i].v, false, bfr[j].v, (short)0, acc[i][j], false, false);
  }
#pragma unroll
  for (int i = 0; i < 4; i++) {
#pragma unroll
    for (int j = 0; j < 2; j++) {
      int col = n0 + wn * 32 + j * 16 + lrow;
      float bb = b2[(size_t)e * D + col];
#pragma unroll
      for (int r = 0; r < 8; r++) {
        int row = m0 + wm * 64 + i * 16 + r + half * 8;
        out2[((size_t)e * CAP + row) * D + col] = acc[i][j][r] + bb;
      }
    }
  }
}

// ---------------- combine: y[s] = sum_k w_k*out2[...] + (sum_k w_k)*x[s] ----------------
__global__ __launch_bounds__(256)
void moe_combine(const float* __restrict__ x, const int* __restrict__ topk_idx,
                 const float* __restrict__ topk_val, const int* __restrict__ pos,
                 const float* __restrict__ out2, float* __restrict__ y) {
  int s = blockIdx.x;
  int a0 = s * 2, a1 = s * 2 + 1;
  int e0 = topk_idx[a0], e1 = topk_idx[a1];
  int p0 = pos[a0], p1 = pos[a1];
  bool k0 = (p0 < CAP), k1 = (p1 < CAP);
  float w0 = k0 ? topk_val[a0] : 0.0f;
  float w1 = k1 ? topk_val[a1] : 0.0f;
  const float* r0 = out2 + ((size_t)e0 * CAP + (k0 ? p0 : 0)) * D;
  const float* r1 = out2 + ((size_t)e1 * CAP + (k1 ? p1 : 0)) * D;
  const float* xr = x + (size_t)s * D;
  float* yr = y + (size_t)s * D;
  float ws = w0 + w1;
  int d = threadIdx.x * 4;
  float4 o0; o0.x = o0.y = o0.z = o0.w = 0.0f;
  float4 o1; o1.x = o1.y = o1.z = o1.w = 0.0f;
  if (k0) o0 = *reinterpret_cast<const float4*>(&r0[d]);
  if (k1) o1 = *reinterpret_cast<const float4*>(&r1[d]);
  float4 xv = *reinterpret_cast<const float4*>(&xr[d]);
  float4 o;
  o.x = w0 * o0.x + w1 * o1.x + ws * xv.x;
  o.y = w0 * o0.y + w1 * o1.y + ws * xv.y;
  o.z = w0 * o0.z + w1 * o1.z + ws * xv.z;
  o.w = w0 * o0.w + w1 * o1.w + ws * xv.w;
  *reinterpret_cast<float4*>(&yr[d]) = o;
}

__global__ void moe_aux(const float* __restrict__ imp, const float* __restrict__ cnt,
                        float* __restrict__ out_aux) {
  if (threadIdx.x == 0 && blockIdx.x == 0) {
    float ssum = 0.0f;
    for (int e = 0; e < E; e++)
      ssum += (imp[e] / (float)S) * (cnt[e] / (float)S);
    *out_aux = (float)E * ssum;
  }
}

// ---------------- launch ----------------
extern "C" void kernel_launch(void* const* d_in, const int* in_sizes, int n_in,
                              void* d_out, int out_size, void* d_ws, size_t ws_size,
                              hipStream_t stream) {
  (void)in_sizes; (void)n_in; (void)out_size; (void)ws_size;
  const float* x  = (const float*)d_in[0];
  const float* wg = (const float*)d_in[1];
  const float* W1 = (const float*)d_in[2];
  const float* b1 = (const float*)d_in[3];
  const float* W2 = (const float*)d_in[4];
  const float* b2 = (const float*)d_in[5];

  char* ws = (char*)d_ws;
  float*  imp  = (float*)(ws + OFF_IMP);
  float*  cnt  = (float*)(ws + OFF_CNT);
  int*    idx  = (int*)(ws + OFF_IDX);
  float*  val  = (float*)(ws + OFF_VAL);
  int*    pos  = (int*)(ws + OFF_POS);
  bf16_t* W1b  = (bf16_t*)(ws + OFF_W1B);
  bf16_t* W2b  = (bf16_t*)(ws + OFF_W2B);
  bf16_t* bufb = (bf16_t*)(ws + OFF_BUFB);
  bf16_t* H    = (bf16_t*)(ws + OFF_H);
  float*  out2 = (float*)(ws + OFF_OUT2);

  float* y   = (float*)d_out;
  float* aux = y + (size_t)S * D;

  zero_f32_kernel<<<1, 32, 0, stream>>>(imp, E);
  moe_router<<<S / 8, 256, 0, stream>>>(x, wg, idx, val, imp);
  moe_scan<<<1, 256, 0, stream>>>(idx, pos, cnt);

  size_t nzero = (size_t)E * CAP * D / 8; // uint4 units (8 bf16 each)
  zero_b128_kernel<<<(unsigned)((nzero + 255) / 256), 256, 0, stream>>>((uint4*)bufb, nzero);
  moe_dispatch<<<SK, 256, 0, stream>>>(x, idx, pos, bufb);

  size_t n4w = (size_t)E * F * D / 4; // float4 units
  f32_to_bf16_kernel<<<(unsigned)((n4w + 255) / 256), 256, 0, stream>>>(W1, W1b, n4w);
  f32_to_bf16_kernel<<<(unsigned)((n4w + 255) / 256), 256, 0, stream>>>(W2, W2b, n4w);

  moe_gemm1<<<dim3(CAP / 128, F / 128, E), 256, 0, stream>>>(bufb, W1b, b1, H);
  moe_gemm2<<<dim3(CAP / 128, D / 128, E), 256, 0, stream>>>(H, W2b, b2, out2);

  moe_combine<<<S, 256, 0, stream>>>(x, idx, val, pos, out2, y);
  moe_aux<<<1, 32, 0, stream>>>(imp, cnt, aux);
}
